// Attention_52295521796784
// MI455X (gfx1250) — matrix operand order repaired and checked by reading, not compiled, not run on hardware
//
#include <hip/hip_runtime.h>

// ---------------------------------------------------------------------------
// MHA forward for B=4, N=2048, C=512, 8 heads x d=64 on gfx1250 (CDNA5).
// All matmuls on V_WMMA_F32_16X16X32_BF16 (f32 accumulate).
//   K0: one-shot fp32 -> bf16 conversion of x / qkv weights / proj weights
//   K1: fused QKV projection, 16x64 tile/wave, pure-bf16 operand loads
//   K2: flash attention, 16 q-rows/wave, 64-key tiles, online softmax
//   K3: output projection + bias, 16x64 tile/wave, pure-bf16 operand loads
// ---------------------------------------------------------------------------

#define BATCH 4
#define SEQ   2048
#define CH    512
#define HEADS 8
#define HD    64

typedef __bf16 bf16_t;
typedef bf16_t v16bf __attribute__((ext_vector_type(16)));
typedef float  v8f   __attribute__((ext_vector_type(8)));

// A/B operand fragment: 16 bf16 values per lane, viewable as 8 dwords.
union FragAB {
    v16bf        v;
    unsigned int u[8];
};

// float -> bf16 bits, round-to-nearest-even
__device__ __forceinline__ unsigned int bf_bits(float f) {
    unsigned int u = __float_as_uint(f);
    u += 0x7FFFu + ((u >> 16) & 1u);
    return u >> 16;
}
// pack two floats into a bf16x2 dword (lo = first element)
__device__ __forceinline__ unsigned int pack_bf2(float lo, float hi) {
    return bf_bits(lo) | (bf_bits(hi) << 16);
}

__device__ __forceinline__ v8f wmma_bf16(const FragAB& a, const FragAB& b, v8f c) {
    // (neg_a, A, neg_b, B, c_mod, C, reuse_a, reuse_b)
    return __builtin_amdgcn_wmma_f32_16x16x32_bf16(false, a.v, false, b.v,
                                                   (short)0, c, false, false);
}

// A-fragment k order used by this code (contiguous, not AMD's table order): lane m = lane&15, half kh = lane>>4.
// VGPR v : K = kh*16 + 2v  (pairs)
// -> dwords v0..7 are one contiguous 32-byte run per lane.
__device__ __forceinline__ int a_koff(int v, int kh) {
    return kh * 16 + 2 * v;
}
// B-fragment k order used by this code (same as A, not AMD's table order): lane n = lane&15, VGPR v: K = kh*16 + 2v (pairs)
// -> dwords v0..7 are one contiguous 32-byte run per lane.

// Load a 16x32 A fragment from a row-major bf16 row pointer (elements, +kk)
__device__ __forceinline__ FragAB load_a(const unsigned short* row, int kk, int kh) {
    FragAB a;
    #pragma unroll
    for (int v = 0; v < 8; ++v)
        a.u[v] = *(const unsigned int*)(row + kk + a_koff(v, kh));
    return a;
}
// Load a 32x16 B fragment whose column data is contiguous along k
__device__ __forceinline__ FragAB load_b(const unsigned short* col, int kh) {
    FragAB b;
    #pragma unroll
    for (int v = 0; v < 8; ++v)
        b.u[v] = *(const unsigned int*)(col + kh * 16 + 2 * v);
    return b;
}

// ---------------------------------------------------------------------------
// Kernel 0: elementwise fp32 -> bf16 (8 elements per thread, b128 in/out)
// ---------------------------------------------------------------------------
__global__ __launch_bounds__(256) void cvt_bf16_kernel(
        const float* __restrict__ src, unsigned short* __restrict__ dst, int n8)
{
    const int i = blockIdx.x * 256 + threadIdx.x;
    if (i < n8) {
        const float4* s = (const float4*)src + (size_t)i * 2;
        const float4 lo = s[0], hi = s[1];
        uint4 o;
        o.x = pack_bf2(lo.x, lo.y);
        o.y = pack_bf2(lo.z, lo.w);
        o.z = pack_bf2(hi.x, hi.y);
        o.w = pack_bf2(hi.z, hi.w);
        *((uint4*)dst + i) = o;
    }
}

// ---------------------------------------------------------------------------
// Kernel 1: fused QKV projection, 16(M) x 64(F) tile per wave, bf16 operands.
//   Xb [8192 x 512] x Wqkv [1536 x 512] (rows = out features) -> Q/K/V bf16
//   Q: [b,h,n,d] pre-scaled by 1/8 (exact).  K: [b,h,n,d].  V: [b,h,d,n].
// 64-wide feature tile is wave-uniformly Q, K or V (one head).
// ---------------------------------------------------------------------------
__global__ __launch_bounds__(128) void qkv_proj_kernel(
        const unsigned short* __restrict__ Xb, const unsigned short* __restrict__ Wqkv,
        unsigned short* __restrict__ Qb, unsigned short* __restrict__ Kb,
        unsigned short* __restrict__ Vt)
{
    const int lane = threadIdx.x & 31;
    const int wave = threadIdx.x >> 5;
    const int tile = blockIdx.x * 4 + wave;
    const int FT = (3 * CH) / 64;          // 24 feature tiles (64 wide)
    const int mt = tile / FT;
    const int ft = tile % FT;
    const int m0 = mt * 16;
    const int f0 = ft * 64;

    const int lm = lane & 15;
    const int kh = lane >> 4;

    const unsigned short* arow = Xb + (size_t)(m0 + lm) * CH;
    const unsigned short* wbase = Wqkv + (size_t)f0 * CH;

    v8f c[4] = {};
    for (int kk = 0; kk < CH; kk += 32) {
        const FragAB a = load_a(arow, kk, kh);
        #pragma unroll
        for (int t = 0; t < 4; ++t) {
            const FragAB b = load_b(wbase + (size_t)(t * 16 + lm) * CH + kk, kh);
            c[t] = wmma_bf16(a, b, c[t]);
        }
    }

    const float scale = 0.125f;            // HD^-0.5, exact power of two
    if (f0 < CH) {                         // ---- Q tile (head = f0/64)
        const int hh = f0 >> 6;
        #pragma unroll
        for (int r = 0; r < 8; ++r) {
            const int g  = m0 + kh * 8 + r;
            const int bb = g >> 11, nn = g & (SEQ - 1);
            unsigned short* dst =
                Qb + (((size_t)(bb * HEADS + hh) * SEQ) + nn) * HD;
            #pragma unroll
            for (int t = 0; t < 4; ++t)
                dst[t * 16 + lm] = (unsigned short)bf_bits(c[t][r] * scale);
        }
    } else if (f0 < 2 * CH) {              // ---- K tile
        const int hh = (f0 - CH) >> 6;
        #pragma unroll
        for (int r = 0; r < 8; ++r) {
            const int g  = m0 + kh * 8 + r;
            const int bb = g >> 11, nn = g & (SEQ - 1);
            unsigned short* dst =
                Kb + (((size_t)(bb * HEADS + hh) * SEQ) + nn) * HD;
            #pragma unroll
            for (int t = 0; t < 4; ++t)
                dst[t * 16 + lm] = (unsigned short)bf_bits(c[t][r]);
        }
    } else {                               // ---- V tile (store transposed)
        const int hh = (f0 - 2 * CH) >> 6;
        #pragma unroll
        for (int r = 0; r < 8; ++r) {
            const int g  = m0 + kh * 8 + r;
            const int bb = g >> 11, nn = g & (SEQ - 1);
            unsigned short* dst =
                Vt + ((size_t)(bb * HEADS + hh) * HD) * SEQ + nn;
            #pragma unroll
            for (int t = 0; t < 4; ++t)
                dst[(size_t)(t * 16 + lm) * SEQ] = (unsigned short)bf_bits(c[t][r]);
        }
    }
}

// ---------------------------------------------------------------------------
// Kernel 2: flash attention.  1 wave = 16 query rows, 4 waves/block.
// Grid = B*H * (N/64) = 1024 blocks.  64-key tiles: 16 WMMAs per iteration
// (8 for S = Q*K^T, 8 for O += P*V), P staged through LDS as bf16.
// Output written directly as bf16 for the bf16 out-projection GEMM.
// ---------------------------------------------------------------------------
__global__ __launch_bounds__(128) void flash_attn_kernel(
        const unsigned short* __restrict__ Qb, const unsigned short* __restrict__ Kb,
        const unsigned short* __restrict__ Vt, unsigned short* __restrict__ AOb)
{
    __shared__ unsigned short lds_p[4][16 * 64];   // bf16 P staging, 2 KB/wave

    const int lane = threadIdx.x & 31;
    const int wave = threadIdx.x >> 5;
    const int qblk = blockIdx.x & 31;      // N/64 = 32 query blocks
    const int bh   = blockIdx.x >> 5;      // batch*head
    const int b    = bh >> 3;
    const int h    = bh & 7;
    const int q0   = qblk * 64 + wave * 16;

    const int lm = lane & 15;
    const int kh = lane >> 4;

    // Preload Q A-fragments for full head dim (k = 0..31 and 32..63)
    const unsigned short* qrow = Qb + ((size_t)bh * SEQ + q0 + lm) * HD;
    const FragAB qa0 = load_a(qrow, 0, kh);
    const FragAB qa1 = load_a(qrow, 32, kh);

    float mi[8], li[8];
    v8f o[4];
    const v8f vzero = {};
    #pragma unroll
    for (int r = 0; r < 8; ++r) { mi[r] = -1e30f; li[r] = 0.0f; }
    #pragma unroll
    for (int t = 0; t < 4; ++t) o[t] = vzero;

    unsigned short* lsp = lds_p[wave];

    for (int j0 = 0; j0 < SEQ; j0 += 64) {
        // Prefetch next key/value tiles into cache (global_prefetch_b8)
        if (j0 + 64 < SEQ) {
            __builtin_prefetch(Kb + ((size_t)bh * SEQ + j0 + 64 + lane) * HD, 0, 3);
            __builtin_prefetch(Vt + ((size_t)bh * HD + lane * 2) * SEQ + j0 + 64, 0, 3);
        }

        // ---- S = Q * K^T for 4 column tiles of 16 keys each
        v8f s[4];
        #pragma unroll
        for (int cb = 0; cb < 4; ++cb) {
            const unsigned short* kb =
                Kb + ((size_t)bh * SEQ + j0 + cb * 16 + lm) * HD;
            s[cb] = wmma_bf16(qa0, load_b(kb, kh), vzero);
            s[cb] = wmma_bf16(qa1, load_b(kb + 32, kh), s[cb]);
        }

        // ---- online softmax: row r lives across the 16 lanes of this half
        #pragma unroll
        for (int r = 0; r < 8; ++r) {
            float mx = fmaxf(fmaxf(s[0][r], s[1][r]), fmaxf(s[2][r], s[3][r]));
            mx = fmaxf(mx, __shfl_xor(mx, 1, 32));
            mx = fmaxf(mx, __shfl_xor(mx, 2, 32));
            mx = fmaxf(mx, __shfl_xor(mx, 4, 32));
            mx = fmaxf(mx, __shfl_xor(mx, 8, 32));
            const float mnew  = fmaxf(mi[r], mx);
            const float alpha = __expf(mi[r] - mnew);
            mi[r] = mnew;
            float e[4], rs = 0.0f;
            #pragma unroll
            for (int cb = 0; cb < 4; ++cb) { e[cb] = __expf(s[cb][r] - mnew); rs += e[cb]; }
            rs += __shfl_xor(rs, 1, 32);
            rs += __shfl_xor(rs, 2, 32);
            rs += __shfl_xor(rs, 4, 32);
            rs += __shfl_xor(rs, 8, 32);
            li[r] = li[r] * alpha + rs;
            #pragma unroll
            for (int t = 0; t < 4; ++t) o[t][r] *= alpha;
            // stage P as bf16 in C layout (row-major 16x64)
            const int row = kh * 8 + r;
            #pragma unroll
            for (int cb = 0; cb < 4; ++cb)
                lsp[row * 64 + cb * 16 + lm] = (unsigned short)bf_bits(e[cb]);
        }
        __syncthreads();   // fence compiler; HW DS pipe is in-order per wave

        // ---- P A-fragments (k = 0..31, 32..63) straight from LDS dwords
        const FragAB pa0 = load_a(lsp + lm * 64, 0, kh);
        const FragAB pa1 = load_a(lsp + lm * 64, 32, kh);

        // ---- O += P * V ;  V^T stored [bh][d][n]: pairs contiguous along n
        #pragma unroll
        for (int t = 0; t < 4; ++t) {
            const unsigned short* vb =
                Vt + ((size_t)bh * HD + t * 16 + lm) * SEQ + j0;
            o[t] = wmma_bf16(pa0, load_b(vb, kh), o[t]);
            o[t] = wmma_bf16(pa1, load_b(vb + 32, kh), o[t]);
        }
        __syncthreads();   // WAR: keep LDS reads ahead of next tile's stores
    }

    // Epilogue: O / l, write bf16 attn-out in [b,n,c] with c = h*64+d
    #pragma unroll
    for (int r = 0; r < 8; ++r) {
        const int row = kh * 8 + r;
        const float inv = 1.0f / li[r];
        const size_t g = (size_t)b * SEQ + q0 + row;
        #pragma unroll
        for (int t = 0; t < 4; ++t)
            AOb[g * CH + h * HD + t * 16 + lm] =
                (unsigned short)bf_bits(o[t][r] * inv);
    }
}

// ---------------------------------------------------------------------------
// Kernel 3: output projection, 16(M) x 64(F) tile per wave, bf16 operands,
// fp32 bias + fp32 output.
// ---------------------------------------------------------------------------
__global__ __launch_bounds__(128) void out_proj_kernel(
        const unsigned short* __restrict__ AOb, const unsigned short* __restrict__ Wp,
        const float* __restrict__ proj_b, float* __restrict__ out)
{
    const int lane = threadIdx.x & 31;
    const int wave = threadIdx.x >> 5;
    const int tile = blockIdx.x * 4 + wave;
    const int FT = CH / 64;                // 8
    const int mt = tile / FT;
    const int ft = tile % FT;
    const int m0 = mt * 16, f0 = ft * 64;
    const int lm = lane & 15, kh = lane >> 4;

    const unsigned short* arow = AOb + (size_t)(m0 + lm) * CH;
    const unsigned short* wbase = Wp + (size_t)f0 * CH;

    v8f c[4] = {};
    for (int kk = 0; kk < CH; kk += 32) {
        const FragAB a = load_a(arow, kk, kh);
        #pragma unroll
        for (int t = 0; t < 4; ++t) {
            const FragAB b = load_b(wbase + (size_t)(t * 16 + lm) * CH + kk, kh);
            c[t] = wmma_bf16(a, b, c[t]);
        }
    }

    float bias[4];
    #pragma unroll
    for (int t = 0; t < 4; ++t) bias[t] = proj_b[f0 + t * 16 + lm];
    #pragma unroll
    for (int r = 0; r < 8; ++r) {
        const int g = m0 + kh * 8 + r;
        #pragma unroll
        for (int t = 0; t < 4; ++t)
            out[(size_t)g * CH + f0 + t * 16 + lm] = c[t][r] + bias[t];
    }
}

// ---------------------------------------------------------------------------
// Host launch.  Workspace layout (bytes, all bf16 unless noted):
//   [ 0.0M,  8.0M)  Xb    [8192][512]
//   [ 8.0M,  9.5M)  Wqkv  [1536][512]   (q_w rows then kv_w rows)
//   [ 9.5M, 10.0M)  Wp    [512][512]
//   [10.0M, 18.0M)  Qb    [b,h,n,d]     (pre-scaled)
//   [18.0M, 26.0M)  Kb    [b,h,n,d]
//   [26.0M, 34.0M)  Vt    [b,h,d,n]
//   [34.0M, 42.0M)  AOb   [8192][512]
// ---------------------------------------------------------------------------
extern "C" void kernel_launch(void* const* d_in, const int* in_sizes, int n_in,
                              void* d_out, int out_size, void* d_ws, size_t ws_size,
                              hipStream_t stream) {
    const float* x      = (const float*)d_in[0];
    const float* q_w    = (const float*)d_in[1];
    const float* kv_w   = (const float*)d_in[2];
    const float* proj_w = (const float*)d_in[3];
    const float* proj_b = (const float*)d_in[4];
    float* out = (float*)d_out;

    char* ws = (char*)d_ws;
    const size_t MB = 1u << 20;
    unsigned short* Xb   = (unsigned short*)(ws);
    unsigned short* Wqkv = (unsigned short*)(ws + 8 * MB);
    unsigned short* Wp   = (unsigned short*)(ws + 9 * MB + MB / 2);
    unsigned short* Qb   = (unsigned short*)(ws + 10 * MB);
    unsigned short* Kb   = (unsigned short*)(ws + 18 * MB);
    unsigned short* Vt   = (unsigned short*)(ws + 26 * MB);
    unsigned short* AOb  = (unsigned short*)(ws + 34 * MB);

    // 0) one-shot fp32 -> bf16 conversions (8 elems/thread)
    cvt_bf16_kernel<<<(BATCH * SEQ * CH) / 8 / 256, 256, 0, stream>>>(x, Xb,
                                                    (BATCH * SEQ * CH) / 8);
    cvt_bf16_kernel<<<(CH * CH) / 8 / 256, 256, 0, stream>>>(q_w, Wqkv,
                                                    (CH * CH) / 8);
    cvt_bf16_kernel<<<(2 * CH * CH) / 8 / 256, 256, 0, stream>>>(kv_w,
                                  Wqkv + (size_t)CH * CH, (2 * CH * CH) / 8);
    cvt_bf16_kernel<<<(CH * CH) / 8 / 256, 256, 0, stream>>>(proj_w, Wp,
                                                    (CH * CH) / 8);

    // 1) QKV projection: 512 M-tiles x 24 F64-tiles, 4 wave-tiles per block
    qkv_proj_kernel<<<(512 * 24) / 4, 128, 0, stream>>>(Xb, Wqkv, Qb, Kb, Vt);
    // 2) Flash attention: B*H (32) x N/64 (32) blocks
    flash_attn_kernel<<<32 * 32, 128, 0, stream>>>(Qb, Kb, Vt, AOb);
    // 3) Output projection: 512 M-tiles x 8 F64-tiles, 4 wave-tiles per block
    out_proj_kernel<<<(512 * 8) / 4, 128, 0, stream>>>(AOb, Wp, proj_b, out);
}
